// Phi4_20246475834102
// MI455X (gfx1250) — matrix operand order repaired and checked by reading, not compiled, not run on hardware
//
#include <hip/hip_runtime.h>
#include <hip/hip_bf16.h>

// ---------------------------------------------------------------------------
// Phi4 action:  out[b] = x_b^T K x_b  +  lamb * sum_sites (sum_ch x^2)^2
//   x: [256, 4, 64, 64] f32   K: [16384, 16384] f32 (symmetric)   out: [256] f32
// bf16 WMMA GEMM streaming K exactly once (HBM floor ~46us @ 23.3 TB/s).
// X kept bf16 in ws (L2-resident). Fully pipelined:
//   - async global->LDS staging (ASYNCcnt) ping-pongs across 2 LDS buffers,
//   - next B tile (NT f32) prefetched into regs during current chunk's WMMAs,
//   - A tiles preloaded in parity-indexed register groups (no dscnt-0 stalls).
// ---------------------------------------------------------------------------

#define NTOT   16384
#define NBATCH 256
#define ISPLIT 4
#define ICHUNK (NTOT / ISPLIT / 32)   // 128 k-chunks of 32 per block
#define LDS_PITCH 40                  // bf16 elems per row = 80 B (bank-conflict-free)
#define BUFELEMS (NBATCH * LDS_PITCH) // 10240 bf16 = 20 KB per buffer

typedef __attribute__((ext_vector_type(16))) __bf16 v16bf;
typedef __attribute__((ext_vector_type(8)))  float  v8f;
typedef __attribute__((ext_vector_type(4)))  float  v4f;

// ---- CDNA5 async global->LDS copy (per-lane 16B), tracked by ASYNCcnt ------
__device__ __forceinline__ void async_g2l_b128(unsigned lds_byte_addr, const void* g) {
    asm volatile("global_load_async_to_lds_b128 %0, %1, off"
                 :: "v"(lds_byte_addr), "v"(g) : "memory");
}
__device__ __forceinline__ void wait_asynccnt0() {
#if __has_builtin(__builtin_amdgcn_s_wait_asynccnt)
    __builtin_amdgcn_s_wait_asynccnt(0);
#else
    asm volatile("s_wait_asynccnt 0x0" ::: "memory");
#endif
}

// ---- Kernel 1: convert X f32 -> bf16 into workspace ------------------------
__global__ __launch_bounds__(256) void phi4_cvt_kernel(const float* __restrict__ x,
                                                       __bf16* __restrict__ xb) {
    int i = (blockIdx.x * 256 + threadIdx.x) * 4;
    v4f f = *(const v4f*)(x + i);
    union { __bf16 e[4]; uint2 q; } u;
    u.e[0] = (__bf16)f[0]; u.e[1] = (__bf16)f[1];
    u.e[2] = (__bf16)f[2]; u.e[3] = (__bf16)f[3];
    *(uint2*)(xb + i) = u.q;
}

// ---- Kernel 2: quartic term, also initializes d_out ------------------------
__global__ __launch_bounds__(256) void phi4_quart_kernel(const float* __restrict__ x,
                                                         float* __restrict__ out) {
    const int b = blockIdx.x;
    const float* xb = x + (size_t)b * NTOT;     // [4][4096] channel-major
    float s = 0.f;
    for (int site = threadIdx.x; site < 4096; site += 256) {
        float a0 = xb[site];
        float a1 = xb[site + 4096];
        float a2 = xb[site + 8192];
        float a3 = xb[site + 12288];
        float q = a0 * a0 + a1 * a1 + a2 * a2 + a3 * a3;
        s += q * q;
    }
    for (int off = 16; off > 0; off >>= 1) s += __shfl_xor(s, off, 32);
    __shared__ float red[8];
    if ((threadIdx.x & 31) == 0) red[threadIdx.x >> 5] = s;
    __syncthreads();
    if (threadIdx.x == 0) {
        float t = 0.f;
        for (int w = 0; w < 8; ++w) t += red[w];
        out[b] = 0.02f * t;                      // LAMB = 0.02, init output
    }
}

// ---- Kernel 3: quad = x^T K x via bf16 WMMA --------------------------------
// grid (128, ISPLIT), block 256 (8 waves). Wave w owns 16 columns of K and all
// 256 batch rows; blockIdx.y splits the contraction 4-way (K still read once).
__global__ __launch_bounds__(256) void phi4_gemm_kernel(const float* __restrict__ Km,
                                                        const float* __restrict__ X,
                                                        const __bf16* __restrict__ Xb,
                                                        float* __restrict__ out) {
    __shared__ __bf16 As[2][BUFELEMS];          // 2 x 20 KB ping-pong

    const int lane  = threadIdx.x & 31;
    const int wave  = threadIdx.x >> 5;
    const int nlo   = lane & 15;                // column within 16-wide N tile
    const int hi    = lane >> 4;                // lane half (0/1)
    const int j16   = blockIdx.x * 128 + wave * 16;
    const int ibase = blockIdx.y * (NTOT / ISPLIT);
    const int aoff  = hi * 16;                  // A: K 0..15 vs 16..31

    v8f acc[16];
    #pragma unroll
    for (int t = 0; t < 16; ++t)
        #pragma unroll
        for (int v = 0; v < 8; ++v) acc[t][v] = 0.f;

    union BV { v16bf v; __bf16 e[16]; };
    union AV { v16bf v; uint4  q[2]; };

    const size_t krow_base = (size_t)(j16 + nlo) * NTOT;   // symmetric access
    const int my_row = threadIdx.x;                        // one LDS row per thread
    const unsigned lds0 = (unsigned)(size_t)&As[0][my_row * LDS_PITCH];
    const unsigned lds1 = (unsigned)(size_t)&As[1][my_row * LDS_PITCH];
    const char* grow = (const char*)(Xb + (size_t)my_row * NTOT);

    // ---- prologue: stage chunk 0 into buf0, prefetch B(0) ------------------
    {
        const char* g = grow + (size_t)ibase * 2;
        async_g2l_b128(lds0,      g);
        async_g2l_b128(lds0 + 16, g + 16);
        async_g2l_b128(lds0 + 32, g + 32);
        async_g2l_b128(lds0 + 48, g + 48);
    }
    const v4f* kp0 = (const v4f*)(Km + krow_base + ibase + hi * 16);
    v4f f0 = __builtin_nontemporal_load(kp0 + 0);
    v4f f1 = __builtin_nontemporal_load(kp0 + 1);
    v4f f2 = __builtin_nontemporal_load(kp0 + 2);
    v4f f3 = __builtin_nontemporal_load(kp0 + 3);
    wait_asynccnt0();
    __syncthreads();

    for (int ic = 0; ic < ICHUNK; ++ic) {
        const int cur = ic & 1;
        const int icn = (ic + 1 < ICHUNK) ? (ic + 1) : (ICHUNK - 1);
        const int i0n = ibase + icn * 32;

        // --- issue async staging of NEXT chunk into the other buffer --------
        {
            const unsigned l = cur ? lds0 : lds1;
            const char* g = grow + (size_t)i0n * 2;
            async_g2l_b128(l,      g);
            async_g2l_b128(l + 16, g + 16);
            async_g2l_b128(l + 32, g + 32);
            async_g2l_b128(l + 48, g + 48);
        }

        // --- convert the prefetched B tile (32x16 KxN, via K symmetry) ------
        BV bv;
        #pragma unroll
        for (int q = 0; q < 4; ++q) {
            bv.e[q]      = (__bf16)f0[q];
            bv.e[4 + q]  = (__bf16)f1[q];
            bv.e[8 + q]  = (__bf16)f2[q];
            bv.e[12 + q] = (__bf16)f3[q];
        }

        // --- prefetch NEXT B tile (NT: don't pollute L2 with the K stream) --
        const v4f* kp = (const v4f*)(Km + krow_base + i0n + hi * 16);
        f0 = __builtin_nontemporal_load(kp + 0);
        f1 = __builtin_nontemporal_load(kp + 1);
        f2 = __builtin_nontemporal_load(kp + 2);
        f3 = __builtin_nontemporal_load(kp + 3);

        // --- 16 batch tiles, A preloaded in parity-indexed register groups --
        const __bf16* abase = &As[cur][nlo * LDS_PITCH];
        AV a[2][2];
        #pragma unroll
        for (int u = 0; u < 2; ++u) {           // preload tiles 0,1 -> a[0]
            const __bf16* arow = abase + u * 16 * LDS_PITCH;
            a[0][u].q[0] = *(const uint4*)(arow + aoff);
            a[0][u].q[1] = *(const uint4*)(arow + aoff + 8);
        }
        #pragma unroll
        for (int g = 0; g < 8; ++g) {
            if (g < 7) {                        // preload group g+1 -> a[(g+1)&1]
                #pragma unroll
                for (int u = 0; u < 2; ++u) {
                    const __bf16* arow = abase + ((g + 1) * 2 + u) * 16 * LDS_PITCH;
                    a[(g + 1) & 1][u].q[0] = *(const uint4*)(arow + aoff);
                    a[(g + 1) & 1][u].q[1] = *(const uint4*)(arow + aoff + 8);
                }
            }
            acc[2 * g + 0] = __builtin_amdgcn_wmma_f32_16x16x32_bf16(
                false, a[g & 1][0].v, false, bv.v, (short)0, acc[2 * g + 0], false, false);
            acc[2 * g + 1] = __builtin_amdgcn_wmma_f32_16x16x32_bf16(
                false, a[g & 1][1].v, false, bv.v, (short)0, acc[2 * g + 1], false, false);
        }

        // --- close the chunk: our staging done, everyone synced -------------
        wait_asynccnt0();
        __syncthreads();
    }

    // Epilogue: out[b] += sum_n X[b][j16+n] * Y[b][j16+n]
    // C layout: VGPR v -> row M = v (lanes 0-15) or v+8 (lanes 16-31), N = nlo.
    #pragma unroll
    for (int t = 0; t < 16; ++t) {
        #pragma unroll
        for (int v = 0; v < 8; ++v) {
            const int brow = t * 16 + v + hi * 8;
            float val = acc[t][v] * X[(size_t)brow * NTOT + j16 + nlo];
            val += __shfl_xor(val, 1, 32);
            val += __shfl_xor(val, 2, 32);
            val += __shfl_xor(val, 4, 32);
            val += __shfl_xor(val, 8, 32);
            if (nlo == 0) atomicAdd(&out[brow], val);
        }
    }
}

// ---------------------------------------------------------------------------
extern "C" void kernel_launch(void* const* d_in, const int* in_sizes, int n_in,
                              void* d_out, int out_size, void* d_ws, size_t ws_size,
                              hipStream_t stream) {
    const float* x  = (const float*)d_in[0];   // 256*4*64*64 = 4194304 f32
    const float* Km = (const float*)d_in[1];   // 16384*16384 f32
    float* out = (float*)d_out;                // 256 f32
    __bf16* xb = (__bf16*)d_ws;                // 8 MiB bf16 copy of x

    phi4_cvt_kernel<<<dim3((NBATCH * NTOT) / (256 * 4)), dim3(256), 0, stream>>>(x, xb);
    phi4_quart_kernel<<<dim3(NBATCH), dim3(256), 0, stream>>>(x, out);
    phi4_gemm_kernel<<<dim3(NTOT / 128, ISPLIT), dim3(256), 0, stream>>>(Km, x, xb, out);
}
